// _MockNonStandardAttention_71519795413050
// MI455X (gfx1250) — matrix operand order repaired and checked by reading, not compiled, not run on hardware
//
#include <hip/hip_runtime.h>

typedef __bf16 bf16;
typedef __attribute__((ext_vector_type(16))) __bf16 v16bf;
typedef __attribute__((ext_vector_type(8)))  float   v8f;
typedef int v4i __attribute__((vector_size(16)));

#if defined(__has_builtin)
#  if __has_builtin(__builtin_amdgcn_global_load_async_to_lds_b128)
#    define HAVE_ASYNC_LDS 1
#  else
#    define HAVE_ASYNC_LDS 0
#  endif
#  if __has_builtin(__builtin_amdgcn_s_wait_asynccnt)
#    define WAIT_ASYNC() __builtin_amdgcn_s_wait_asynccnt(0)
#  else
#    define WAIT_ASYNC() asm volatile("s_wait_asynccnt 0" ::: "memory")
#  endif
#else
#  define HAVE_ASYNC_LDS 0
#  define WAIT_ASYNC() asm volatile("s_wait_asynccnt 0" ::: "memory")
#endif

#if HAVE_ASYNC_LDS
// Build AS1/AS3 int4* by casting through integers (avoids const / addrspace
// cast legality issues). AS3 pointer value = low 32 bits of generic LDS addr.
#define GPTR(p) ((__attribute__((address_space(1))) v4i*)(unsigned long long)(const void*)(p))
#define LPTR(p) ((__attribute__((address_space(3))) v4i*)(unsigned int)(unsigned long long)(const void*)(p))
#endif

static __device__ __forceinline__ v8f wmma_bf16(v16bf a, v16bf b, v8f c) {
  // D = A(16x32 bf16) * B(32x16 bf16) + C(16x16 f32)
  return __builtin_amdgcn_wmma_f32_16x16x32_bf16(false, a, false, b, (short)0, c, false, false);
}

static __device__ __forceinline__ v8f vzero8() {
  v8f v = {0.f, 0.f, 0.f, 0.f, 0.f, 0.f, 0.f, 0.f};
  return v;
}

static __device__ __forceinline__ unsigned pack2(float x, float y) {
  return (unsigned)__builtin_bit_cast(unsigned short, (bf16)x) |
         ((unsigned)__builtin_bit_cast(unsigned short, (bf16)y) << 16);
}

// A-fragment gather: 16x32 bf16 tile, row-major, `stride` elements per row.
// ISA layout: lanes 0-15 (row=lane): VGPR j -> K = {0,2,4,6,16,18,20,22}[j]..+1
//             lanes 16-31 (row=lane-16): K shifted by +8.
static __device__ __forceinline__ v16bf load_a_frag(const bf16* base, int stride, int lane) {
  const int half = lane >> 4;
  const int m    = lane & 15;
  v16bf f;
#pragma unroll
  for (int j = 0; j < 8; ++j) {
    const int k0 = (j < 4 ? 2 * j : 2 * j + 8) + 8 * half;
    unsigned u = *(const unsigned*)(base + m * stride + k0);
    f[2 * j]     = __builtin_bit_cast(bf16, (unsigned short)(u & 0xffffu));
    f[2 * j + 1] = __builtin_bit_cast(bf16, (unsigned short)(u >> 16));
  }
  return f;
}

// B-fragment gather from an N-major tile: tile[n*stride + k], n=0..15, k=0..31.
// ISA layout: lanes 0-15 (col=lane): VGPR j -> K = {0,2,4,6,16,18,20,22}[j]..+1; lanes 16-31: K shifted by +8.
static __device__ __forceinline__ v16bf load_b_frag(const bf16* base, int stride, int lane) {
  const int half = lane >> 4;
  const int n    = lane & 15;
  v16bf f;
#pragma unroll
  for (int j = 0; j < 8; ++j) {
    const int k0 = (j < 4 ? 2 * j : 2 * j + 8) + 8 * half;
    unsigned u = *(const unsigned*)(base + n * stride + k0);
    f[2 * j]     = __builtin_bit_cast(bf16, (unsigned short)(u & 0xffffu));
    f[2 * j + 1] = __builtin_bit_cast(bf16, (unsigned short)(u >> 16));
  }
  return f;
}

// ---------------------------------------------------------------------------
// GEMM: C(M x N) = A(M x 1024) * Bw(1024 x N), bf16 WMMA, fp32 accumulate.
// Block: 128 threads (4 waves). Tile 128(M) x 128(N) x 32(K). Wave tile 64x64
// (16 accumulators -> 16 WMMA per K-step vs 8 fragment gathers, 4:1 DS ratio).
// MODE 0: A is fp32 (hidden_states); epilogue scatters Q/K (B,H,S,D) and
//         V transposed (B,H,D,S) as bf16 into workspace.
// MODE 1: A is bf16 (attention output); epilogue writes fp32 to d_out.
// ---------------------------------------------------------------------------
template <bool A_IS_F32, int MODE>
__global__ __launch_bounds__(128) void gemm_bf16_wmma(const void* __restrict__ A_,
                                                      const float* __restrict__ Bw,
                                                      void* __restrict__ o0,
                                                      void* __restrict__ o1,
                                                      void* __restrict__ o2,
                                                      int N) {
  constexpr int K = 1024;
  const int Mbase = blockIdx.y * 128;
  const int Nbase = blockIdx.x * 128;

  __shared__ bf16 lA[128 * 34];  // 128 rows x 32 K (stride 34 keeps pairs 4B aligned)
  __shared__ bf16 lB[128 * 34];  // N-major: 128 n-rows x 32 K

  const int tid  = threadIdx.x;
  const int w    = tid >> 5;
  const int lane = tid & 31;
  const int wr   = w >> 1;  // 0..1 -> 64-row slice
  const int wc   = w & 1;   // 0..1 -> 64-col slice

  v8f acc[4][4];
#pragma unroll
  for (int r = 0; r < 4; ++r)
#pragma unroll
    for (int c = 0; c < 4; ++c) acc[r][c] = vzero8();

  for (int kt = 0; kt < K / 32; ++kt) {
    __syncthreads();
    // ---- Stage A tile (128 x 32) ----
    if (A_IS_F32) {
      // 1024 float4 chunks / 128 threads = 8 each
#pragma unroll
      for (int i = 0; i < 8; ++i) {
        const int c = tid + 128 * i;             // 0..1023
        const int r = c >> 3, cp = c & 7;        // row, 4-float chunk
        const float4 f =
            ((const float4*)A_)[((Mbase + r) * K + kt * 32 + cp * 4) >> 2];
        unsigned* dst = (unsigned*)&lA[r * 34 + cp * 4];
        dst[0] = pack2(f.x, f.y);
        dst[1] = pack2(f.z, f.w);
      }
    } else {
      // bf16 A: 512 16B chunks / 128 threads = 4 each
#pragma unroll
      for (int i = 0; i < 4; ++i) {
        const int c = tid + 128 * i;             // 0..511
        const int r = c >> 2, cp = c & 3;        // row, 8-elem chunk
        const uint4 u =
            *(const uint4*)((const bf16*)A_ + (Mbase + r) * K + kt * 32 + cp * 8);
        unsigned* dst = (unsigned*)&lA[r * 34 + cp * 8];
        dst[0] = u.x; dst[1] = u.y; dst[2] = u.z; dst[3] = u.w;
      }
    }
    // ---- Stage B tile (32K x 128N), transposed into N-major layout ----
#pragma unroll
    for (int i = 0; i < 8; ++i) {
      const int c = tid + 128 * i;               // 0..1023
      const int kk = c >> 5, np = c & 31;        // k-row, 4-n chunk
      const float4 f =
          ((const float4*)Bw)[((kt * 32 + kk) * N + Nbase + np * 4) >> 2];
      lB[(np * 4 + 0) * 34 + kk] = (bf16)f.x;
      lB[(np * 4 + 1) * 34 + kk] = (bf16)f.y;
      lB[(np * 4 + 2) * 34 + kk] = (bf16)f.z;
      lB[(np * 4 + 3) * 34 + kk] = (bf16)f.w;
    }
    __syncthreads();

    v16bf af[4];
#pragma unroll
    for (int r = 0; r < 4; ++r)
      af[r] = load_a_frag(&lA[(64 * wr + 16 * r) * 34], 34, lane);
#pragma unroll
    for (int c = 0; c < 4; ++c) {
      const v16bf bfrag = load_b_frag(&lB[(64 * wc + 16 * c) * 34], 34, lane);
#pragma unroll
      for (int r = 0; r < 4; ++r) acc[r][c] = wmma_bf16(af[r], bfrag, acc[r][c]);
    }
  }

  // Epilogue. C layout: VGPR i -> row (i, lanes 0-15) / (i+8, lanes 16-31); col = lane&15.
  const int half = lane >> 4;
  const int nl   = lane & 15;
#pragma unroll
  for (int r = 0; r < 4; ++r)
#pragma unroll
    for (int c = 0; c < 4; ++c)
#pragma unroll
      for (int i = 0; i < 8; ++i) {
        const int gm = Mbase + 64 * wr + 16 * r + i + 8 * half;  // 0..4095 (b*S+s)
        const int gn = Nbase + 64 * wc + 16 * c + nl;
        const float v = acc[r][c][i];
        if (MODE == 0) {
          const int b = gm >> 11, s = gm & 2047;
          const int sel = gn >> 10, nn = gn & 1023;
          const int h = nn >> 6, d = nn & 63;
          const int bh = (b << 4) + h;
          const bf16 bv = (bf16)v;
          if (sel == 0)
            ((bf16*)o0)[(bh * 2048 + s) * 64 + d] = bv;  // Q: (B,H,S,D)
          else if (sel == 1)
            ((bf16*)o1)[(bh * 2048 + s) * 64 + d] = bv;  // K: (B,H,S,D)
          else
            ((bf16*)o2)[(bh * 64 + d) * 2048 + s] = bv;  // V: (B,H,D,S)
        } else {
          ((float*)o0)[gm * N + gn] = v;
        }
      }
}

// ---------------------------------------------------------------------------
// Causal flash attention, bf16 WMMA, fp32 online softmax.
// Block: 128 threads (4 waves), one 64-query tile per block; wave owns 16 rows.
// Grid: B*H*(S/64) = 1024. Key blocks of 32 iterate up to the causal bound.
// K/V blocks staged with GLOBAL_LOAD_ASYNC_TO_LDS_B128 when available.
// ---------------------------------------------------------------------------
__global__ __launch_bounds__(128) void attn_flash_wmma(const bf16* __restrict__ qb,
                                                       const bf16* __restrict__ kbuf,
                                                       const bf16* __restrict__ vb,
                                                       bf16* __restrict__ attn) {
  const int bid = blockIdx.x;
  const int qt  = bid & 31;        // S/64 = 32 query tiles
  const int bh  = bid >> 5;        // b*16 + h
  const int b   = bh >> 4;
  const int h   = bh & 15;

  // Strides chosen so rows are 16B-aligned (async B128 LDS writes) and even
  // (4B-aligned bf16-pair fragment gathers).
  __shared__ bf16 lK[32 * 72];        // 32 keys x 64 dim (row-major)
  __shared__ bf16 lV[64 * 40];        // 64 dim x 32 keys (N-major for PV)
  __shared__ bf16 lP[4 * 16 * 34];    // per-wave 16x32 probs scratch

  const int tid  = threadIdx.x;
  const int w    = tid >> 5;
  const int lane = tid & 31;
  const int half = lane >> 4;
  const int nl   = lane & 15;
  const int qrow0 = qt * 64 + w * 16;

  // Q fragments held in registers for the whole kernel (head dim 64 = 2 k-steps).
  const bf16* qptr = qb + (bh * 2048 + qrow0) * 64;
  const v16bf qf0 = load_a_frag(qptr, 64, lane);
  const v16bf qf1 = load_a_frag(qptr + 32, 64, lane);

  float mrow[8], lrow[8];
  v8f oacc[4];
#pragma unroll
  for (int i = 0; i < 8; ++i) { mrow[i] = -1e30f; lrow[i] = 0.f; }
#pragma unroll
  for (int t = 0; t < 4; ++t) oacc[t] = vzero8();

  const float scale = 0.125f;           // 64^-0.5
  const int nkb = (qt * 64 + 64) >> 5;  // causal bound: keys < qtile end

  for (int kb = 0; kb < nkb; ++kb) {
    __syncthreads();
#if HAVE_ASYNC_LDS
    // K tile: 256 16B chunks; V tile: 256 16B chunks; 2+2 per thread.
#pragma unroll
    for (int i = 0; i < 2; ++i) {
      const int c = tid + 128 * i;                 // 0..255
      const int key = c >> 3, dp = c & 7;          // K: 8 chunks per key row
      __builtin_amdgcn_global_load_async_to_lds_b128(
          GPTR(kbuf + (bh * 2048 + kb * 32 + key) * 64 + dp * 8),
          LPTR(&lK[key * 72 + dp * 8]), 0, 0);
      const int d = c >> 2, kp = c & 3;            // V: 4 chunks per dim row
      __builtin_amdgcn_global_load_async_to_lds_b128(
          GPTR(vb + (bh * 64 + d) * 2048 + kb * 32 + kp * 8),
          LPTR(&lV[d * 40 + kp * 8]), 0, 0);
    }
    WAIT_ASYNC();
#else
#pragma unroll
    for (int i = 0; i < 16; ++i) {
      const int e = tid + 128 * i;                 // 0..2047
      const int key = e >> 6, d = e & 63;
      lK[key * 72 + d] = kbuf[(bh * 2048 + kb * 32 + key) * 64 + d];
      const int d2 = e >> 5, key2 = e & 31;
      lV[d2 * 40 + key2] = vb[(bh * 64 + d2) * 2048 + kb * 32 + key2];
    }
#endif
    __syncthreads();

    // scores S(16 x 32) = Q(16x64) * K^T : two 16x16 accumulators, 2 k-steps each
    v8f sacc[2];
#pragma unroll
    for (int t = 0; t < 2; ++t) {
      sacc[t] = vzero8();
      const v16bf kf0 = load_b_frag(&lK[16 * t * 72], 72, lane);
      const v16bf kf1 = load_b_frag(&lK[16 * t * 72 + 32], 72, lane);
      sacc[t] = wmma_bf16(qf0, kf0, sacc[t]);
      sacc[t] = wmma_bf16(qf1, kf1, sacc[t]);
    }

    // Online softmax (row stats replicated across each 16-lane half).
#pragma unroll
    for (int i = 0; i < 8; ++i) {
      const int qrow = qrow0 + i + 8 * half;
      const int key0 = kb * 32 + nl;
      float s0 = sacc[0][i] * scale;
      float s1 = sacc[1][i] * scale;
      if (key0 > qrow) s0 = -1e30f;
      if (key0 + 16 > qrow) s1 = -1e30f;

      float r = fmaxf(s0, s1);
      r = fmaxf(r, __shfl_xor(r, 1, 16));
      r = fmaxf(r, __shfl_xor(r, 2, 16));
      r = fmaxf(r, __shfl_xor(r, 4, 16));
      r = fmaxf(r, __shfl_xor(r, 8, 16));

      const float mn = fmaxf(mrow[i], r);
      const float corr = __expf(mrow[i] - mn);
      const float p0 = __expf(s0 - mn);
      const float p1 = __expf(s1 - mn);
      float ps = p0 + p1;
      ps += __shfl_xor(ps, 1, 16);
      ps += __shfl_xor(ps, 2, 16);
      ps += __shfl_xor(ps, 4, 16);
      ps += __shfl_xor(ps, 8, 16);

      lrow[i] = lrow[i] * corr + ps;
      mrow[i] = mn;
#pragma unroll
      for (int t = 0; t < 4; ++t) oacc[t][i] = oacc[t][i] * corr;

      // Spill probs to per-wave LDS scratch (C layout -> row-major tile).
      bf16* pp = &lP[w * 544 + (i + 8 * half) * 34];
      pp[nl]      = (bf16)p0;
      pp[16 + nl] = (bf16)p1;
    }

    // In-wave DS ordering: our stores must land before the A-layout gather.
    asm volatile("s_wait_dscnt 0" ::: "memory");

    const v16bf pf = load_a_frag(&lP[w * 544], 34, lane);
#pragma unroll
    for (int t = 0; t < 4; ++t) {
      const v16bf vf = load_b_frag(&lV[16 * t * 40], 40, lane);
      oacc[t] = wmma_bf16(pf, vf, oacc[t]);
    }
  }

  // Normalize and store to (B, S, H*D) bf16.
  float inv[8];
#pragma unroll
  for (int i = 0; i < 8; ++i) inv[i] = 1.0f / lrow[i];
#pragma unroll
  for (int t = 0; t < 4; ++t)
#pragma unroll
    for (int i = 0; i < 8; ++i) {
      const int qrow = qrow0 + i + 8 * half;
      attn[(b * 2048 + qrow) * 1024 + h * 64 + 16 * t + nl] = (bf16)(oacc[t][i] * inv[i]);
    }
}

// ---------------------------------------------------------------------------
extern "C" void kernel_launch(void* const* d_in, const int* in_sizes, int n_in,
                              void* d_out, int out_size, void* d_ws, size_t ws_size,
                              hipStream_t stream) {
  (void)in_sizes; (void)n_in; (void)out_size; (void)ws_size;
  const float* x     = (const float*)d_in[0];  // (2,2048,1024)
  const float* w_qkv = (const float*)d_in[1];  // (1024,3072)
  const float* w_out = (const float*)d_in[2];  // (1024,1024)
  float*       out   = (float*)d_out;          // (2,2048,1024)

  const size_t QKV_ELEMS = (size_t)2 * 16 * 2048 * 64;  // 4,194,304
  bf16* qbuf = (bf16*)d_ws;
  bf16* kbuf = qbuf + QKV_ELEMS;
  bf16* vbuf = kbuf + QKV_ELEMS;
  bf16* attn = vbuf + QKV_ELEMS;                        // (4096,1024) bf16

  // 1) QKV projection + layout scatter (V stored transposed).
  gemm_bf16_wmma<true, 0><<<dim3(3072 / 128, 4096 / 128), 128, 0, stream>>>(
      x, w_qkv, qbuf, kbuf, vbuf, 3072);

  // 2) Causal flash attention.
  attn_flash_wmma<<<dim3(2 * 16 * 32), 128, 0, stream>>>(qbuf, kbuf, vbuf, attn);

  // 3) Output projection -> fp32.
  gemm_bf16_wmma<false, 1><<<dim3(1024 / 128, 4096 / 128), 128, 0, stream>>>(
      attn, w_out, out, nullptr, nullptr, 1024);
}
